// ConverterSurrogate_20452634264259
// MI455X (gfx1250) — matrix operand order repaired and checked by reading, not compiled, not run on hardware
//
#include <hip/hip_runtime.h>
#include <hip/hip_bf16.h>

// ---------------------------------------------------------------------------
// 2-layer LSTM (B=256, T=1000, H=128, D_in=3) + 3-wide FC head, for gfx1250.
// 16 persistent blocks x 16 batch rows; weights LDS-resident as bf16;
// per-step recurrent GEMM via v_wmma_f32_16x16x32_bf16; f32 cell state;
// per-step 16x128 bf16 tile traffic moved by the Tensor Data Mover
// (tensor_load_to_lds / tensor_store_from_lds, TENSORcnt), double-buffered
// in layer 1 so the DMA for step t+1 overlaps step t's GEMM + gate math.
// ---------------------------------------------------------------------------

#define B_   256
#define T_   1000
#define DIN  3
#define H_   128
#define G4   512   // 4*H
#define MB   16    // batch rows per block

typedef __attribute__((ext_vector_type(16))) __bf16    v16bf;
typedef __attribute__((ext_vector_type(8)))  __bf16    v8bf;
typedef __attribute__((ext_vector_type(8)))  float     v8f;
typedef __attribute__((ext_vector_type(4)))  unsigned  v4u;
typedef __attribute__((ext_vector_type(8)))  int       v8i;
typedef __attribute__((ext_vector_type(4)))  int       v4i;

#if __has_builtin(__builtin_amdgcn_tensor_load_to_lds) && \
    __has_builtin(__builtin_amdgcn_tensor_store_from_lds)
#define HAVE_TDM 1
#else
#define HAVE_TDM 0
#endif

union AF { v16bf v; v8bf h[2]; };

__device__ __forceinline__ float sigf(float x) { return 1.0f / (1.0f + __expf(-x)); }

#if HAVE_TDM
typedef const __attribute__((address_space(3))) void* lds_cptr;

__device__ __forceinline__ unsigned lds_off(const void* p) {
  // LDS (addrspace 3) pointers are 32-bit byte offsets into the WG's LDS.
  return (unsigned)(unsigned long long)(lds_cptr)p;
}

// D# groups 0/1 for a 16x128 bf16 tile: rows contiguous in LDS (256B each),
// global rows strided by T_*H_ elements (row-per-batch layout of h1[B,T,H]).
__device__ __forceinline__ void tdm_desc_16x128(unsigned lds_byte,
                                                const void* gaddr,
                                                v4u& g0, v8i& g1) {
  unsigned long long ga = (unsigned long long)gaddr;
  g0[0] = 1u;                                    // count=1, user descriptor
  g0[1] = lds_byte;                              // lds_addr (bytes)
  g0[2] = (unsigned)(ga & 0xffffffffu);          // global_addr[31:0]
  g0[3] = (unsigned)((ga >> 32) & 0x01ffffffu)   // global_addr[56:32]
        | (2u << 30);                            // type=2 ("image")
  g1[0] = (int)(1u << 16);                       // data_size=1 -> 2 bytes/elem
  g1[1] = (int)((unsigned)H_ << 16);             // tensor_dim0 = 128
  g1[2] = (int)((unsigned)MB << 16);             // tensor_dim1 = 16
  g1[3] = (int)((unsigned)H_ << 16);             // tile_dim0 = 128
  g1[4] = MB;                                    // tile_dim1 = 16
  g1[5] = (int)(T_ * H_);                        // tensor_dim0_stride = 128000
  g1[6] = 0;                                     // stride hi / dim1_stride lo
  g1[7] = 0;
}

__device__ __forceinline__ void tdm_load(v4u g0, v8i g1) {
  v4i z4 = {0, 0, 0, 0};
#if defined(__clang_major__) && (__clang_major__ >= 23)
  v8i z8 = {0, 0, 0, 0, 0, 0, 0, 0};
  __builtin_amdgcn_tensor_load_to_lds(g0, g1, z4, z4, z8, 0);
#else
  __builtin_amdgcn_tensor_load_to_lds(g0, g1, z4, z4, 0);
#endif
}

__device__ __forceinline__ void tdm_store(v4u g0, v8i g1) {
  v4i z4 = {0, 0, 0, 0};
#if defined(__clang_major__) && (__clang_major__ >= 23)
  v8i z8 = {0, 0, 0, 0, 0, 0, 0, 0};
  __builtin_amdgcn_tensor_store_from_lds(g0, g1, z4, z4, z8, 0);
#else
  __builtin_amdgcn_tensor_store_from_lds(g0, g1, z4, z4, 0);
#endif
}

__device__ __forceinline__ void tdm_wait0() {
#if __has_builtin(__builtin_amdgcn_s_wait_tensorcnt)
  __builtin_amdgcn_s_wait_tensorcnt(0);
#else
  asm volatile("s_wait_tensorcnt 0x0" ::: "memory");
#endif
}
#endif  // HAVE_TDM

// ---------------------------------------------------------------------------
// Layer 0: x (B,T,3) -> h1 (B,T,128) bf16 in workspace.
// LDS (bytes): Wb[512*128]bf16 @0 (131072) | gbuf[16*512]f32 @131072 (32768)
//   cbuf[16*128]f32 @163840 (8192) | hA[16*128]bf16 @172032 (4096)
//   wih[512*3]f32 @176128 (6144) | bias[512]f32 @182272 (2048)
//   xbuf[16*4]f32 @184320 (256)   -> total 184576
// ---------------------------------------------------------------------------
__global__ __launch_bounds__(256) void lstm_l0(
    const float* __restrict__ x,   const float* __restrict__ Wih,
    const float* __restrict__ Whh, const float* __restrict__ bih,
    const float* __restrict__ bhh, __bf16* __restrict__ h1g)
{
  extern __shared__ char smem[];
  __bf16* Wb   = (__bf16*)(smem);
  float*  gbuf = (float*) (smem + 131072);
  float*  cbuf = (float*) (smem + 163840);
  __bf16* hA   = (__bf16*)(smem + 172032);
  float*  wih  = (float*) (smem + 176128);
  float*  bias = (float*) (smem + 182272);
  float*  xbuf = (float*) (smem + 184320);

  const int tid = threadIdx.x;
  const int b0  = blockIdx.x * MB;

  for (int i = tid; i < G4 * H_;  i += 256) Wb[i]  = (__bf16)Whh[i];
  for (int i = tid; i < G4 * DIN; i += 256) wih[i] = Wih[i];
  for (int i = tid; i < G4;       i += 256) bias[i] = bih[i] + bhh[i];
  for (int i = tid; i < MB * H_;  i += 256) { hA[i] = (__bf16)0.0f; cbuf[i] = 0.0f; }
  __syncthreads();

  const int wave = tid >> 5, lane = tid & 31;
  const int half = lane >> 4, mrow = lane & 15, ncol = lane & 15;

  for (int t = 0; t < T_; ++t) {
    if (tid < MB * DIN) {                 // stage x[b0..b0+15, t, 0..2]
      int r = tid / 3, i = tid - 3 * r;
      xbuf[r * 4 + i] = x[((size_t)(b0 + r) * T_ + t) * DIN + i];
    }
#if HAVE_TDM
    // DMA previous step's h tile (still intact in hA) out to h1; overlaps
    // this step's GEMM. Waited for before hA is overwritten (below).
    if (wave == 0 && t > 0) {
      v4u g0; v8i g1;
      tdm_desc_16x128(lds_off(hA), h1g + ((size_t)b0 * T_ + (t - 1)) * H_, g0, g1);
      tdm_store(g0, g1);
    }
#endif
    __syncthreads();

    // A fragments: h_prev (16x128 bf16), shared across all N tiles.
    AF a[4];
#pragma unroll
    for (int kt = 0; kt < 4; ++kt) {
      const __bf16* ap = hA + mrow * H_ + kt * 32 + half * 8;
      a[kt].h[0] = *(const v8bf*)ap;
      a[kt].h[1] = *(const v8bf*)(ap + 16);
    }

    // g = h_prev @ W_hh^T  (M=16, N=512, K=128): 4 N-tiles per wave.
#pragma unroll
    for (int nt = 0; nt < 4; ++nt) {
      const int n = (wave * 4 + nt) * 16 + ncol;
      v8f acc = {0.f, 0.f, 0.f, 0.f, 0.f, 0.f, 0.f, 0.f};
#pragma unroll
      for (int kt = 0; kt < 4; ++kt) {
        AF b;
        const __bf16* bp = Wb + n * H_ + kt * 32 + half * 8;  // B[k][n]=W[n][k]
        b.h[0] = *(const v8bf*)bp;
        b.h[1] = *(const v8bf*)(bp + 16);
        acc = __builtin_amdgcn_wmma_f32_16x16x32_bf16(
            false, a[kt].v, false, b.v, (short)0, acc, false, false);
      }
      const float w0 = wih[n * 3 + 0], w1 = wih[n * 3 + 1], w2 = wih[n * 3 + 2];
      const float bn = bias[n];
#pragma unroll
      for (int v = 0; v < 8; ++v) {
        const int m = v + half * 8;
        gbuf[m * G4 + n] = acc[v] + bn + xbuf[m * 4 + 0] * w0
                                       + xbuf[m * 4 + 1] * w1
                                       + xbuf[m * 4 + 2] * w2;
      }
    }
#if HAVE_TDM
    if (wave == 0) tdm_wait0();   // t-1 store done before hA is overwritten
#endif
    __syncthreads();

    // Gate nonlinearities + state update (2048 elems / 256 threads).
#pragma unroll
    for (int j = 0; j < 8; ++j) {
      const int idx = tid + j * 256;
      const int m = idx >> 7, hh = idx & 127;
      const float gi = gbuf[m * G4 + hh];
      const float gf = gbuf[m * G4 + H_ + hh];
      const float gg = gbuf[m * G4 + 2 * H_ + hh];
      const float go = gbuf[m * G4 + 3 * H_ + hh];
      const float c  = sigf(gf) * cbuf[m * H_ + hh] + sigf(gi) * tanhf(gg);
      const float h  = sigf(go) * tanhf(c);
      cbuf[m * H_ + hh] = c;
      hA[m * H_ + hh]   = (__bf16)h;
#if !HAVE_TDM
      h1g[((size_t)(b0 + m) * T_ + t) * H_ + hh] = (__bf16)h;
#endif
    }
    __syncthreads();
  }
#if HAVE_TDM
  if (wave == 0) {                       // flush final step's tile
    v4u g0; v8i g1;
    tdm_desc_16x128(lds_off(hA), h1g + ((size_t)b0 * T_ + (T_ - 1)) * H_, g0, g1);
    tdm_store(g0, g1);
    tdm_wait0();
  }
#endif
}

// ---------------------------------------------------------------------------
// Layer 1 + FC head: h1 (B,T,128 bf16) -> y (B,T,3 f32).
// Concatenated GEMM: g = [h1_t ; h_prev] (16x256) @ [W_ih|W_hh]^T (256x512).
// A matrix split: hX ping-pong (TDM destination) + hH (recurrent half).
// LDS (bytes): Wb[512*256]bf16 @0 (262144) | gbuf @262144 (32768)
//   cbuf @294912 (8192) | hX0 @303104 (4096) | hX1 @307200 (4096)
//   hH @311296 (4096) | bias @315392 (2048) | fcw @317440 (1536)
//   fcb @318976 (16)  -> total 318992  (< 320KB WGP limit)
// ---------------------------------------------------------------------------
__global__ __launch_bounds__(256) void lstm_l1_fc(
    const __bf16* __restrict__ h1g, const float* __restrict__ Wih,
    const float* __restrict__ Whh,  const float* __restrict__ bih,
    const float* __restrict__ bhh,  const float* __restrict__ fcwg,
    const float* __restrict__ fcbg, float* __restrict__ y)
{
  extern __shared__ char smem[];
  __bf16* Wb   = (__bf16*)(smem);
  float*  gbuf = (float*) (smem + 262144);
  float*  cbuf = (float*) (smem + 294912);
  __bf16* hX0  = (__bf16*)(smem + 303104);
  __bf16* hX1  = (__bf16*)(smem + 307200);
  __bf16* hH   = (__bf16*)(smem + 311296);
  float*  bias = (float*) (smem + 315392);
  float*  fcw  = (float*) (smem + 317440);
  float*  fcb  = (float*) (smem + 318976);

  const int tid = threadIdx.x;
  const int b0  = blockIdx.x * MB;

  for (int i = tid; i < G4 * 256; i += 256) {   // concat [W_ih | W_hh] as bf16
    int n = i >> 8, k = i & 255;
    Wb[i] = (__bf16)((k < H_) ? Wih[n * H_ + k] : Whh[n * H_ + (k - H_)]);
  }
  for (int i = tid; i < G4;      i += 256) bias[i] = bih[i] + bhh[i];
  for (int i = tid; i < 3 * H_;  i += 256) fcw[i] = fcwg[i];
  if (tid < 3) fcb[tid] = fcbg[tid];
  for (int i = tid; i < MB * H_; i += 256) { hH[i] = (__bf16)0.0f; cbuf[i] = 0.0f; }
  __syncthreads();

  const int wave = tid >> 5, lane = tid & 31;
  const int half = lane >> 4, mrow = lane & 15, ncol = lane & 15;

#if HAVE_TDM
  if (wave == 0) {                      // prime the pipeline: tile for t=0
    v4u g0; v8i g1;
    tdm_desc_16x128(lds_off(hX0), h1g + (size_t)b0 * T_ * H_, g0, g1);
    tdm_load(g0, g1);
  }
#endif

  for (int t = 0; t < T_; ++t) {
    __bf16* hXcur = (t & 1) ? hX1 : hX0;
#if HAVE_TDM
    if (wave == 0) tdm_wait0();         // tile t landed in hXcur
#else
    {  // manual staging fallback (coalesced b128 loads) + next-step prefetch
      const int m = tid >> 4, ch = tid & 15;
      const __bf16* src = h1g + ((size_t)(b0 + m) * T_ + t) * H_ + ch * 8;
      *(v8bf*)(hXcur + m * H_ + ch * 8) = *(const v8bf*)src;
      if (t + 1 < T_) __builtin_prefetch(src + H_, 0, 0);  // global_prefetch_b8
    }
#endif
    __syncthreads();
#if HAVE_TDM
    if (wave == 0 && t + 1 < T_) {      // DMA t+1 overlaps this step's math
      __bf16* hXnext = ((t + 1) & 1) ? hX1 : hX0;
      v4u g0; v8i g1;
      tdm_desc_16x128(lds_off(hXnext), h1g + ((size_t)b0 * T_ + (t + 1)) * H_, g0, g1);
      tdm_load(g0, g1);
    }
#endif

    AF a[8];
#pragma unroll
    for (int kt = 0; kt < 8; ++kt) {
      const __bf16* base = (kt < 4) ? (hXcur + mrow * H_ + kt * 32)
                                    : (hH   + mrow * H_ + (kt - 4) * 32);
      const __bf16* ap = base + half * 8;
      a[kt].h[0] = *(const v8bf*)ap;
      a[kt].h[1] = *(const v8bf*)(ap + 16);
    }
#pragma unroll
    for (int nt = 0; nt < 4; ++nt) {
      const int n = (wave * 4 + nt) * 16 + ncol;
      v8f acc = {0.f, 0.f, 0.f, 0.f, 0.f, 0.f, 0.f, 0.f};
#pragma unroll
      for (int kt = 0; kt < 8; ++kt) {
        AF b;
        const __bf16* bp = Wb + n * 256 + kt * 32 + half * 8;
        b.h[0] = *(const v8bf*)bp;
        b.h[1] = *(const v8bf*)(bp + 16);
        acc = __builtin_amdgcn_wmma_f32_16x16x32_bf16(
            false, a[kt].v, false, b.v, (short)0, acc, false, false);
      }
      const float bn = bias[n];
#pragma unroll
      for (int v = 0; v < 8; ++v) {
        const int m = v + half * 8;
        gbuf[m * G4 + n] = acc[v] + bn;
      }
    }
    __syncthreads();

#pragma unroll
    for (int j = 0; j < 8; ++j) {
      const int idx = tid + j * 256;
      const int m = idx >> 7, hh = idx & 127;
      const float gi = gbuf[m * G4 + hh];
      const float gf = gbuf[m * G4 + H_ + hh];
      const float gg = gbuf[m * G4 + 2 * H_ + hh];
      const float go = gbuf[m * G4 + 3 * H_ + hh];
      const float c  = sigf(gf) * cbuf[m * H_ + hh] + sigf(gi) * tanhf(gg);
      const float h  = sigf(go) * tanhf(c);
      cbuf[m * H_ + hh] = c;
      hH[m * H_ + hh]   = (__bf16)h;    // h_prev for next step's A
    }
    __syncthreads();

    // FC head: y[b,t,o] = fc_b[o] + sum_j h2[b,j]*fc_w[o,j]  (48 lanes busy;
    // stragglers are absorbed at the next step's first barrier).
    if (tid < MB * 3) {
      const int m = tid / 3, o = tid - 3 * m;
      float acc = fcb[o];
#pragma unroll 4
      for (int jj = 0; jj < H_; ++jj)
        acc += (float)hH[m * H_ + jj] * fcw[o * H_ + jj];
      y[((size_t)(b0 + m) * T_ + t) * 3 + o] = acc;
    }
  }
}

extern "C" void kernel_launch(void* const* d_in, const int* in_sizes, int n_in,
                              void* d_out, int out_size, void* d_ws, size_t ws_size,
                              hipStream_t stream) {
  (void)in_sizes; (void)n_in; (void)out_size; (void)ws_size;
  const float* x    = (const float*)d_in[0];
  const float* Wih0 = (const float*)d_in[1];
  const float* Whh0 = (const float*)d_in[2];
  const float* bih0 = (const float*)d_in[3];
  const float* bhh0 = (const float*)d_in[4];
  const float* Wih1 = (const float*)d_in[5];
  const float* Whh1 = (const float*)d_in[6];
  const float* bih1 = (const float*)d_in[7];
  const float* bhh1 = (const float*)d_in[8];
  const float* fcw  = (const float*)d_in[9];
  const float* fcb  = (const float*)d_in[10];
  float*  y  = (float*)d_out;
  __bf16* h1 = (__bf16*)d_ws;          // B*T*H bf16 = 65,536,000 bytes

  constexpr size_t LDS0 = 184576;
  constexpr size_t LDS1 = 318992;

  hipFuncSetAttribute(reinterpret_cast<const void*>(lstm_l0),
                      hipFuncAttributeMaxDynamicSharedMemorySize, (int)LDS0);
  hipFuncSetAttribute(reinterpret_cast<const void*>(lstm_l1_fc),
                      hipFuncAttributeMaxDynamicSharedMemorySize, (int)LDS1);

  lstm_l0<<<dim3(B_ / MB), dim3(256), LDS0, stream>>>(x, Wih0, Whh0, bih0, bhh0, h1);
  lstm_l1_fc<<<dim3(B_ / MB), dim3(256), LDS1, stream>>>(h1, Wih1, Whh1, bih1, bhh1,
                                                         fcw, fcb, y);
}
